// Score_m_50706383896654
// MI455X (gfx1250) — matrix operand order repaired and checked by reading, not compiled, not run on hardware
//
#include <hip/hip_runtime.h>
#include <hip/hip_bf16.h>

typedef _Float16 f16;
typedef __attribute__((ext_vector_type(16))) _Float16 v16h;
typedef __attribute__((ext_vector_type(8)))  _Float16 v8h;
typedef __attribute__((ext_vector_type(8)))  float    v8f;

#define DEV __device__ __forceinline__

DEV float gelu_f(float x){ return 0.5f*x*(1.0f+erff(x*0.7071067811865475f)); }
DEV float sigm_f(float x){ return 1.0f/(1.0f+expf(-x)); }

// CDNA5 WMMA: D = A(16x32 f16) * B(32x16 f16) + C(16x16 f32)
DEV v8f wmma16(v16h a, v16h b, v8f c){
  return __builtin_amdgcn_wmma_f32_16x16x32_f16(false, a, false, b, (short)0, c, false, false);
}

// A fragment (16x32 f16, ISA 7.12.2): per lane, K indices are two contiguous
// 8-half runs: kbase..kbase+7 and kbase+16..kbase+23, kbase = (lane<16 ? 0 : 8).
// rowptr points at element [m][0] of a K-innermost (stride 32) tile row.
DEV v16h load_a_frag(const f16* rowptr, int lane){
  int kb = (lane < 16) ? 0 : 8;
  v8h lo = *(const v8h*)(rowptr + kb);
  v8h hi = *(const v8h*)(rowptr + kb + 16);
  return __builtin_shufflevector(lo, hi, 0,1,2,3,4,5,6,7,8,9,10,11,12,13,14,15);
}
// B fragment (32x16 f16): per lane, K = kbase..kbase+7 and kbase+16..kbase+23, kbase = (lane<16?0:8).
// rowptr points at element [n][0] of a K-innermost (stride 32) tile row.
DEV v16h load_b_frag(const f16* rowptr, int lane){
  int kb = (lane < 16) ? 0 : 8;
  v8h lo = *(const v8h*)(rowptr + kb);
  v8h hi = *(const v8h*)(rowptr + kb + 16);
  return __builtin_shufflevector(lo, hi, 0,1,2,3,4,5,6,7,8,9,10,11,12,13,14,15);
}

// monotone float<->uint encoding for atomic max
DEV unsigned f2ord(float x){ unsigned u = __float_as_uint(x); return (u & 0x80000000u) ? ~u : (u | 0x80000000u); }
DEV float ord2f(unsigned u){ return (u & 0x80000000u) ? __uint_as_float(u ^ 0x80000000u) : __uint_as_float(~u); }

DEV int tokf(int trans, int g, int l){ return trans ? (l*192 + g) : (g*192 + l); }

// ---------------------------------------------------------------- utility
__global__ void fill_u32_kernel(unsigned* p, unsigned v, int n){
  int i = blockIdx.x*blockDim.x + threadIdx.x;
  if (i < n) p[i] = v;
}

// ---------------------------------------------------------------- generic WMMA GEMM
// C[M,N] = act(A[M,K] @ B + bias[N]) + res[M,N]
// B layout: bt=0 -> B[k*N+n], bt=1 -> B[n*K+k]. act: 0 none, 1 relu, 2 gelu.
__global__ void gemm_kernel(const float* __restrict__ A, const float* __restrict__ B,
                            const float* __restrict__ bias, const float* __restrict__ res,
                            float* __restrict__ C, int M, int N, int K, int bt, int act)
{
  __shared__ f16 sA[64*32] __attribute__((aligned(32)));  // [m][k]
  __shared__ f16 sB[32*32] __attribute__((aligned(32)));  // [n][k]
  int tid = threadIdx.x, lane = tid & 31, wave = tid >> 5; // 128 thr, 4 waves
  int m0 = blockIdx.y * 64, n0 = blockIdx.x * 32;
  v8f acc0 = {}, acc1 = {};
  for (int kc = 0; kc < K; kc += 32) {
    if (m0 + 64 <= M && kc + 32 <= K) {
      for (int i = tid; i < 64*32; i += 128) {
        int mm = i >> 5, kk = i & 31;
        sA[i] = (f16)A[(size_t)(m0+mm)*K + kc + kk];
      }
    } else {
      for (int i = tid; i < 64*32; i += 128) {
        int mm = i >> 5, kk = i & 31;
        int gm = m0 + mm, gk = kc + kk;
        sA[i] = (gm < M && gk < K) ? (f16)A[(size_t)gm*K + gk] : (f16)0.f;
      }
    }
    for (int i = tid; i < 32*32; i += 128) {
      int nn = i >> 5, kk = i & 31;
      int gk = kc + kk, gn = n0 + nn;
      float v = 0.f;
      if (gk < K && gn < N) v = bt ? B[(size_t)gn*K + gk] : B[(size_t)gk*N + gn];
      sB[nn*32 + kk] = (f16)v;
    }
    __syncthreads();
    int am = lane & 15;
    v16h a  = load_a_frag(&sA[(wave*16 + am)*32], lane);
    v16h b0 = load_b_frag(&sB[am*32], lane);
    v16h b1 = load_b_frag(&sB[(16 + am)*32], lane);
    acc0 = wmma16(a, b0, acc0);
    acc1 = wmma16(a, b1, acc1);
    __syncthreads();
  }
#pragma unroll
  for (int t = 0; t < 2; ++t) {
#pragma unroll
    for (int r = 0; r < 8; ++r) {
      int m = m0 + wave*16 + r + ((lane >> 4) << 3);
      int n = n0 + t*16 + (lane & 15);
      if (m < M && n < N) {
        float v = (t == 0) ? acc0[r] : acc1[r];
        if (bias) v += bias[n];
        if (act == 1) v = v > 0.f ? v : 0.f;
        else if (act == 2) v = gelu_f(v);
        if (res) v += res[(size_t)m*N + n];
        C[(size_t)m*N + n] = v;
      }
    }
  }
}

// ---------------------------------------------------------------- graph front-end
__global__ void node_score_base_kernel(const float* mie_mee, const float* mie_feat,
                                       const float* w_root, const float* sag_b,
                                       float* node, float* score)
{
  int n = blockIdx.x*blockDim.x + threadIdx.x;
  if (n >= 384) return;
  float s = sag_b[0];
  for (int c = 0; c < 77; ++c) {
    float v = (c < 32) ? mie_mee[c*384 + n] : mie_feat[n*45 + (c-32)];
    node[n*77 + c] = v;
    s += v * w_root[c];
  }
  score[n] = s;
}
__global__ void score_edge_kernel(const float* node, const int* adj, const float* w_rel, float* score)
{
  int e = blockIdx.x*blockDim.x + threadIdx.x;
  if (e >= 12288) return;
  int src = adj[2*e], dst = adj[2*e+1];
  float s = 0.f;
  for (int c = 0; c < 77; ++c) s += node[src*77 + c] * w_rel[c];
  atomicAdd(&score[dst], s);
}
__global__ void rank_kernel(const float* score, int* perm, float* topval, int* new_idx)
{
  int i = blockIdx.x*blockDim.x + threadIdx.x;
  if (i >= 384) return;
  float si = score[i];
  int r = 0;
  for (int j = 0; j < 384; ++j) {
    float sj = score[j];
    if (sj > si || (sj == si && j < i)) ++r;
  }
  new_idx[i] = (r < 192) ? r : -1;
  if (r < 192) { perm[r] = i; topval[r] = si; }
}
__global__ void gather_kernel(const float* node, const int* perm, const float* topval, float* xp)
{
  int r = blockIdx.x, c = threadIdx.x;
  if (c >= 77) return;
  xp[r*77 + c] = node[perm[r]*77 + c] * tanhf(topval[r]);
}
__global__ void remap_edges_kernel(const int* adj, const int* new_idx, int* s2, int* d2, int* maskb)
{
  int e = blockIdx.x*blockDim.x + threadIdx.x;
  if (e >= 12288) return;
  int a = new_idx[adj[2*e]], b = new_idx[adj[2*e+1]];
  int m = (a >= 0 && b >= 0);
  s2[e] = m ? a : 0; d2[e] = m ? b : 0; maskb[e] = m;
}

// ---------------------------------------------------------------- GATv2
__global__ void gat_logits_kernel(const float* xl, const float* xr, const int* s2, const int* d2,
                                  const int* maskb, const float* att, int ch, float* lg, unsigned* mx)
{
  int idx = blockIdx.x*blockDim.x + threadIdx.x;
  if (idx >= 12288*8) return;
  int e = idx >> 3, h = idx & 7;
  int s = s2[e], d = d2[e];
  int hd = 8*ch;
  float acc = 0.f;
  for (int c = 0; c < ch; ++c) {
    float v = xl[s*hd + h*ch + c] + xr[d*hd + h*ch + c];
    v = v > 0.f ? v : 0.2f*v;
    acc += att[h*ch + c] * v;
  }
  float l = maskb[e] ? acc : -1e30f;
  lg[idx] = l;
  atomicMax(&mx[d*8 + h], f2ord(l));
}
__global__ void gat_accum_kernel(const float* xl, const int* s2, const int* d2, const int* maskb,
                                 const float* lg, const unsigned* mx, int ch, float* den, float* num)
{
  int idx = blockIdx.x*blockDim.x + threadIdx.x;
  if (idx >= 12288*8) return;
  int e = idx >> 3, h = idx & 7;
  int d = d2[e], s = s2[e];
  int hd = 8*ch;
  float m = ord2f(mx[d*8 + h]);
  if (!(m > -3.0e38f)) m = 0.f;  // isfinite replacement (-inf -> 0)
  float ex = expf(lg[idx] - m) * (maskb[e] ? 1.f : 0.f);
  atomicAdd(&den[d*8 + h], ex);
  for (int c = 0; c < ch; ++c)
    atomicAdd(&num[d*hd + h*ch + c], ex * xl[s*hd + h*ch + c]);
}
__global__ void gat_final_kernel(const float* num, const float* den, const float* bias,
                                 int ch, int act, float* out)
{
  int hd = 8*ch;
  int i = blockIdx.x*blockDim.x + threadIdx.x;
  if (i >= 192*hd) return;
  int n = i / hd, hc = i % hd, h = hc / ch;
  float v = num[i] / (den[n*8 + h] + 1e-16f) + bias[hc];
  out[i] = act ? gelu_f(v) : v;
}

// ---------------------------------------------------------------- voxel scatter
__global__ void vox_scatter_kernel(const int* vidx, const float* val, float* vox)
{
  int p = blockIdx.x*blockDim.x + threadIdx.x;
  if (p >= 38400) return;
  atomicAdd(&vox[vidx[2*p]*640 + vidx[2*p+1]], val[p]);
}

// ---------------------------------------------------------------- instance-norm-1 statistics
__global__ void in1_stats_mee_kernel(const float* mee_t, float* mean1, float* inv1)
{
  int c = threadIdx.x;
  if (c >= 64) return;
  float s = 0.f, q = 0.f;
  for (int l = 0; l < 384; ++l) { float v = mee_t[l*64 + c]; s += v; q += v*v; }
  float mean = s / 384.f;
  float var  = q / 384.f - mean*mean;
  float iv = rsqrtf(var + 1e-6f);
  mean1[c] = mean;      inv1[c] = iv;       // r1 channels
  mean1[64+c] = mean;   inv1[64+c] = iv;    // r2 channels (same stats)
}
// Gram matrix of mae pixels: G = sum v v^T (32x32), Svec = sum v
__global__ void mae_gram_kernel(const float* mae, float* G, float* Svec)
{
  __shared__ float sv[8*32];
  int lane = threadIdx.x & 31, grp = threadIdx.x >> 5;
  int gid = blockIdx.x*8 + grp;
  int ngrp = gridDim.x*8;
  float g[32];
#pragma unroll
  for (int i = 0; i < 32; ++i) g[i] = 0.f;
  float s = 0.f;
  for (int pix = gid; pix < 147456; pix += ngrp) {
    sv[grp*32 + lane] = mae[(size_t)pix*32 + lane];
    float va = sv[grp*32 + lane];
#pragma unroll
    for (int b = 0; b < 32; ++b) g[b] += va * sv[grp*32 + b];
    s += va;
  }
#pragma unroll
  for (int i = 0; i < 32; ++i) atomicAdd(&G[lane*32 + i], g[i]);
  atomicAdd(&Svec[lane], s);
}
// channel stats of mae_r = W v + b from Gram matrix (analytic)
__global__ void in1_stats_mae_kernel(const float* G, const float* Svec, const float* w,
                                     const float* bias, float* mean1, float* inv1)
{
  int o = threadIdx.x;
  if (o >= 128) return;
  const float N = 147456.f;
  float ws = 0.f;
  for (int a = 0; a < 32; ++a) ws += w[o*32 + a] * Svec[a];
  float quad = 0.f;
  for (int a = 0; a < 32; ++a) {
    float wa = w[o*32 + a];
    for (int b = 0; b < 32; ++b) quad += wa * w[o*32 + b] * G[a*32 + b];
  }
  float bo = bias[o];
  float sum   = ws + N*bo;
  float sumsq = quad + 2.f*bo*ws + N*bo*bo;
  float mean = sum / N;
  float var  = sumsq / N - mean*mean;
  mean1[128 + o] = mean;
  inv1[128 + o]  = rsqrtf(var + 1e-6f);
}

// ---------------------------------------------------------------- stage2: build normed pm on the fly, conv1x1 256->128 (WMMA)
__global__ void stage2_kernel(const float* __restrict__ mae, const float* __restrict__ mee_t,
                              const float* __restrict__ c2d1_w, const float* __restrict__ c2d1_b,
                              const float* __restrict__ mean1, const float* __restrict__ inv1,
                              const float* __restrict__ in1_g, const float* __restrict__ in1_b,
                              const float* __restrict__ c2d2_w, const float* __restrict__ c2d2_b,
                              float* __restrict__ Y)
{
  __shared__ float sMae[64*32];                            // 64 pixels x 32 in-ch
  __shared__ f16 sA[128*32] __attribute__((aligned(32)));  // W2 chunk [o][k]
  __shared__ f16 sB[64*32]  __attribute__((aligned(32)));  // pm-hat [pix][k]
  int tid = threadIdx.x, lane = tid & 31, wave = tid >> 5; // 256 thr, 8 waves
  int pix0 = blockIdx.x * 64;
  for (int i = tid; i < 64*32; i += 256) {
    int p = i >> 5, c = i & 31;
    sMae[i] = mae[(size_t)(pix0 + p)*32 + c];
  }
  __syncthreads();
  v8f acc[4] = {};
  for (int kc = 0; kc < 256; kc += 32) {
    for (int i = tid; i < 128*32; i += 256) {
      int o = i >> 5, kk = i & 31;
      sA[i] = (f16)c2d2_w[o*256 + kc + kk];
    }
    for (int i = tid; i < 64*32; i += 256) {
      int p = i >> 5, kk = i & 31;
      int c = kc + kk;
      int pix = pix0 + p;
      int ii = pix / 384, jj = pix % 384;
      float v;
      if (c < 64)       v = mee_t[ii*64 + c];
      else if (c < 128) v = mee_t[jj*64 + (c - 64)];
      else {
        int o = c - 128;
        float d = c2d1_b[o];
        const float* mv = &sMae[p*32];
        const float* wv = &c2d1_w[o*32];
#pragma unroll
        for (int q = 0; q < 32; ++q) d += wv[q]*mv[q];
        v = d;
      }
      v = gelu_f((v - mean1[c]) * inv1[c] * in1_g[c] + in1_b[c]);
      sB[p*32 + kk] = (f16)v;
    }
    __syncthreads();
    int am = lane & 15;
    v16h a = load_a_frag(&sA[(wave*16 + am)*32], lane);
#pragma unroll
    for (int nt = 0; nt < 4; ++nt) {
      v16h b = load_b_frag(&sB[(nt*16 + am)*32], lane);
      acc[nt] = wmma16(a, b, acc[nt]);
    }
    __syncthreads();
  }
#pragma unroll
  for (int nt = 0; nt < 4; ++nt) {
#pragma unroll
    for (int r = 0; r < 8; ++r) {
      int o = wave*16 + r + ((lane >> 4) << 3);
      int p = pix0 + nt*16 + (lane & 15);
      Y[(size_t)o*147456 + p] = acc[nt][r] + c2d2_b[o];
    }
  }
}

__global__ void in2_stats_kernel(const float* Y, float* mean2, float* inv2)
{
  int c = blockIdx.x;
  __shared__ float ss[256], sq[256];
  float s = 0.f, q = 0.f;
  for (int i = threadIdx.x; i < 147456; i += 256) {
    float v = Y[(size_t)c*147456 + i];
    s += v; q += v*v;
  }
  ss[threadIdx.x] = s; sq[threadIdx.x] = q;
  __syncthreads();
  for (int off = 128; off; off >>= 1) {
    if (threadIdx.x < off) { ss[threadIdx.x] += ss[threadIdx.x+off]; sq[threadIdx.x] += sq[threadIdx.x+off]; }
    __syncthreads();
  }
  if (threadIdx.x == 0) {
    float mean = ss[0] / 147456.f;
    float var  = sq[0] / 147456.f - mean*mean;
    mean2[c] = mean; inv2[c] = rsqrtf(var + 1e-6f);
  }
}

// ---------------------------------------------------------------- stage3: strided 3x3 conv 128->128 (WMMA), output in (h,w,c)
__global__ void stage3_kernel(const float* __restrict__ Y, const float* __restrict__ mean2,
                              const float* __restrict__ inv2, const float* __restrict__ in2_g,
                              const float* __restrict__ in2_b, const float* __restrict__ samp_w,
                              const float* __restrict__ samp_b, float* __restrict__ X)
{
  __shared__ f16 sIn[17*17*32] __attribute__((aligned(32)));  // [spatial][ic] normed+gelu
  __shared__ f16 sW[128*32]    __attribute__((aligned(32)));  // [o][ic] for current tap
  int tid = threadIdx.x, lane = tid & 31, wave = tid >> 5;    // 256 thr, 8 waves
  int oy0 = blockIdx.y*8, ox0 = blockIdx.x*8;
  v8f acc[4] = {};
  for (int cc = 0; cc < 128; cc += 32) {
    for (int i = tid; i < 17*17*32; i += 256) {
      int ic = i & 31, sp = i >> 5;
      int liy = sp / 17, lix = sp % 17;
      int gy = 2*oy0 - 1 + liy, gx = 2*ox0 - 1 + lix;
      float v = 0.f;
      if (gy >= 0 && gy < 384 && gx >= 0 && gx < 384) {
        int c = cc + ic;
        float raw = Y[(size_t)c*147456 + gy*384 + gx];
        v = gelu_f((raw - mean2[c]) * inv2[c] * in2_g[c] + in2_b[c]);
      }
      sIn[i] = (f16)v;
    }
    for (int tap = 0; tap < 9; ++tap) {
      int dy = tap / 3, dx = tap % 3;
      __syncthreads();
      for (int i = tid; i < 128*32; i += 256) {
        int o = i >> 5, ic = i & 31;
        sW[i] = (f16)samp_w[((o*128 + cc + ic)*3 + dy)*3 + dx];
      }
      __syncthreads();
      int am = lane & 15;
      v16h a = load_a_frag(&sW[(wave*16 + am)*32], lane);
#pragma unroll
      for (int nt = 0; nt < 4; ++nt) {
        int n = nt*16 + am;
        int ly = n >> 3, lx = n & 7;
        v16h b = load_b_frag(&sIn[((2*ly + dy)*17 + (2*lx + dx))*32], lane);
        acc[nt] = wmma16(a, b, acc[nt]);
      }
    }
    __syncthreads();
  }
#pragma unroll
  for (int nt = 0; nt < 4; ++nt) {
#pragma unroll
    for (int r = 0; r < 8; ++r) {
      int o = wave*16 + r + ((lane >> 4) << 3);
      int n = nt*16 + (lane & 15);
      int oy = oy0 + (n >> 3), ox = ox0 + (n & 7);
      X[((size_t)(oy*192 + ox))*128 + o] = acc[nt][r] + samp_b[o];
    }
  }
}

// ---------------------------------------------------------------- layer norm (one wave per token)
__global__ void ln_kernel(const float* X, const float* g, const float* b, float* S, int ntok)
{
  int t = blockIdx.x*(blockDim.x >> 5) + (threadIdx.x >> 5);
  int lane = threadIdx.x & 31;
  if (t >= ntok) return;
  const float* x = X + (size_t)t*128;
  float s = 0.f;
  for (int i = lane; i < 128; i += 32) s += x[i];
  for (int off = 16; off; off >>= 1) s += __shfl_xor(s, off, 32);
  float mean = s / 128.f;
  float v = 0.f;
  for (int i = lane; i < 128; i += 32) { float d = x[i]-mean; v += d*d; }
  for (int off = 16; off; off >>= 1) v += __shfl_xor(v, off, 32);
  float inv = rsqrtf(v/128.f + 1e-5f);
  for (int i = lane; i < 128; i += 32) S[(size_t)t*128 + i] = (x[i]-mean)*inv*g[i] + b[i];
}

// ---------------------------------------------------------------- attention (one wave per batch,head,16-row tile)
__global__ void attn_kernel(const float* __restrict__ Q, const float* __restrict__ K,
                            const float* __restrict__ V, float* __restrict__ O, int trans)
{
  __shared__ float sS[16*192];
  __shared__ f16   sP[16*192] __attribute__((aligned(32)));  // [m][k] probabilities
  int lane = threadIdx.x;
  int rt = blockIdx.x, h = blockIdx.y, g = blockIdx.z;
  int am = lane & 15;
  // A fragment from Q: two contiguous 8-float runs per lane
  v16h a;
  {
    int qtok = tokf(trans, g, rt*16 + am);
    const float* qp = &Q[(size_t)qtok*128 + h*32];
    int kb = (lane < 16) ? 0 : 8;
#pragma unroll
    for (int i = 0; i < 8; ++i) { a[i] = (f16)qp[kb + i]; a[i+8] = (f16)qp[kb + 16 + i]; }
  }
  const float scale = 0.17677669529663687f; // 1/sqrt(32)
  for (int ct = 0; ct < 12; ++ct) {
    // B fragment from K^T: two contiguous 8-float runs per lane
    v16h b;
    int ktok = tokf(trans, g, ct*16 + am);
    const float* kp = &K[(size_t)ktok*128 + h*32];
    int kb = (lane < 16) ? 0 : 8;
#pragma unroll
    for (int i = 0; i < 8; ++i) { b[i] = (f16)kp[kb + i]; b[i+8] = (f16)kp[kb + 16 + i]; }
    v8f c = {};
    c = wmma16(a, b, c);
#pragma unroll
    for (int r = 0; r < 8; ++r) {
      int m = r + ((lane >> 4) << 3);
      sS[m*192 + ct*16 + am] = c[r] * scale;
    }
  }
  __syncthreads();
  if (lane < 16) {
    float mx = -1e30f;
    for (int j = 0; j < 192; ++j) mx = fmaxf(mx, sS[lane*192 + j]);
    float s = 0.f;
    for (int j = 0; j < 192; ++j) { float e = expf(sS[lane*192 + j] - mx); sS[lane*192 + j] = e; s += e; }
    float inv = 1.f / s;
    for (int j = 0; j < 192; ++j) sP[lane*192 + j] = (f16)(sS[lane*192 + j] * inv);
  }
  __syncthreads();
  v8f o0 = {}, o1 = {};
  for (int kc = 0; kc < 192; kc += 32) {
    // A fragment from P (LDS, [m][k], row stride 192 halves = 384B, 32B aligned)
    const f16* pp = &sP[am*192 + kc];
    int kb = (lane < 16) ? 0 : 8;
    v8h lo = *(const v8h*)(pp + kb);
    v8h hi = *(const v8h*)(pp + kb + 16);
    v16h pa = __builtin_shufflevector(lo, hi, 0,1,2,3,4,5,6,7,8,9,10,11,12,13,14,15);
    v16h b0, b1;
    int bb = (lane < 16) ? 0 : 8;
#pragma unroll
    for (int hh = 0; hh < 16; ++hh) {
      int vtok = tokf(trans, g, kc + bb + hh + ((hh < 8) ? 0 : 8));
      b0[hh] = (f16)V[(size_t)vtok*128 + h*32 + am];
      b1[hh] = (f16)V[(size_t)vtok*128 + h*32 + 16 + am];
    }
    o0 = wmma16(pa, b0, o0);
    o1 = wmma16(pa, b1, o1);
  }
#pragma unroll
  for (int r = 0; r < 8; ++r) {
    int m = r + ((lane >> 4) << 3);
    int t = tokf(trans, g, rt*16 + m);
    O[(size_t)t*128 + h*32 + am]      = o0[r];
    O[(size_t)t*128 + h*32 + 16 + am] = o1[r];
  }
}

// ---------------------------------------------------------------- final masked reduction
__global__ void final_reduce_kernel(const float* X, const float* rep1d,
                                    const float* mask_w, const float* mask_b, float* partial)
{
  __shared__ float sp[256];
  int pix = blockIdx.x*256 + threadIdx.x;
  float v = 0.f;
  if (pix < 36864) {
    const float* x = X + (size_t)pix*128;
    float m = mask_b[0];
    for (int c = 0; c < 128; ++c) m += x[c]*mask_w[c];
    v = rep1d[pix / 192] * m;
  }
  sp[threadIdx.x] = v;
  __syncthreads();
  for (int off = 128; off; off >>= 1) {
    if (threadIdx.x < off) sp[threadIdx.x] += sp[threadIdx.x + off];
    __syncthreads();
  }
  if (threadIdx.x == 0) atomicAdd(partial, sp[0]);
}
__global__ void final_sigmoid_kernel(const float* partial, float* out)
{
  if (threadIdx.x == 0) out[0] = sigm_f(partial[0] / 36864.f);
}

// ================================================================ host
extern "C" void kernel_launch(void* const* d_in, const int* in_sizes, int n_in,
                              void* d_out, int out_size, void* d_ws, size_t ws_size,
                              hipStream_t stream)
{
  (void)in_sizes; (void)n_in; (void)out_size; (void)ws_size;
  const float* mie_mee  = (const float*)d_in[0];
  const float* mie_feat = (const float*)d_in[1];
  const int*   mae_adj  = (const int*)d_in[2];
  const int*   mee_vidx = (const int*)d_in[3];
  const float* mee_val  = (const float*)d_in[4];
  const float* mae      = (const float*)d_in[5];
#define PAR(i) ((const float*)d_in[i])

  float* W = (float*)d_ws;
  size_t off = 0;
  auto A_ = [&](size_t n){ float* p = W + off; off += (n + 63) & ~(size_t)63; return p; };
  float* node   = A_(384*77);
  float* score  = A_(384);
  int*   perm   = (int*)A_(192);
  float* topval = A_(192);
  int*   newidx = (int*)A_(384);
  int*   s2     = (int*)A_(12288);
  int*   d2     = (int*)A_(12288);
  int*   maskb  = (int*)A_(12288);
  float* xp     = A_(192*77);
  float* xl     = A_(192*128);
  float* xr     = A_(192*128);
  float* lg     = A_(12288*8);
  unsigned* mx  = (unsigned*)A_(192*8);
  float* den    = A_(192*8);
  float* num    = A_(192*128);
  float* xa     = A_(192*128);
  float* xb     = A_(192*128);
  float* rep1d  = A_(192);
  float* vox    = A_(384*640);
  float* mee_t  = A_(384*64);
  float* G      = A_(1024);
  float* Svec   = A_(32);
  float* mean1  = A_(256);
  float* inv1   = A_(256);
  float* mean2  = A_(128);
  float* inv2   = A_(128);
  float* partial= A_(64);
  float* Xpp    = A_((size_t)36864*128);
  // overlay: Y (stage2 out) is dead before the transformer buffers are live
  float* Y  = W + off;                       // 128*147456 f32
  float* Sb = W + off;
  float* Qb = Sb + (size_t)36864*128;
  float* Kb = Qb + (size_t)36864*128;
  float* Vb = Kb + (size_t)36864*128;
  float* Ob = Vb + (size_t)36864*128;
  float* Hb = Ob + (size_t)36864*128;

  auto fill = [&](void* p, unsigned v, int n){
    fill_u32_kernel<<<(n + 255)/256, 256, 0, stream>>>((unsigned*)p, v, n);
  };
  auto gemm = [&](const float* Am, const float* Bm, const float* bias, const float* res,
                  float* Cm, int M, int N, int K, int bt, int act){
    dim3 grid((N + 31)/32, (M + 63)/64);
    gemm_kernel<<<grid, 128, 0, stream>>>(Am, Bm, bias, res, Cm, M, N, K, bt, act);
  };

  // ---- graph front-end
  node_score_base_kernel<<<3, 128, 0, stream>>>(mie_mee, mie_feat, PAR(7), PAR(8), node, score);
  score_edge_kernel<<<48, 256, 0, stream>>>(node, mae_adj, PAR(6), score);
  rank_kernel<<<3, 128, 0, stream>>>(score, perm, topval, newidx);
  gather_kernel<<<192, 96, 0, stream>>>(node, perm, topval, xp);
  remap_edges_kernel<<<48, 256, 0, stream>>>(mae_adj, newidx, s2, d2, maskb);

  // ---- GATv2 x4
  const int WLi[4] = {9,13,17,21}, WRi[4] = {10,14,18,22}, ATi[4] = {11,15,19,23}, BBi[4] = {12,16,20,24};
  const int IND[4] = {77,128,128,128}, CH[4] = {16,16,16,8};
  float* xin = xp;
  float* outs[4] = {xa, xb, xa, xb};
  for (int l = 0; l < 4; ++l) {
    int hd = 8*CH[l];
    gemm(xin, PAR(WLi[l]), nullptr, nullptr, xl, 192, hd, IND[l], 0, 0);
    gemm(xin, PAR(WRi[l]), nullptr, nullptr, xr, 192, hd, IND[l], 0, 0);
    fill(mx, 0x007FFFFFu, 192*8);   // f2ord(-inf)
    fill(den, 0u, 192*8);
    fill(num, 0u, 192*hd);
    gat_logits_kernel<<<384, 256, 0, stream>>>(xl, xr, s2, d2, maskb, PAR(ATi[l]), CH[l], lg, mx);
    gat_accum_kernel<<<384, 256, 0, stream>>>(xl, s2, d2, maskb, lg, mx, CH[l], den, num);
    gat_final_kernel<<<(192*hd + 255)/256, 256, 0, stream>>>(num, den, PAR(BBi[l]), CH[l], (l < 3) ? 1 : 0, outs[l]);
    xin = outs[l];
  }
  gemm(xin, PAR(25), PAR(26), nullptr, rep1d, 192, 1, 64, 0, 0);   // rep1d

  // ---- voxel branch + mee
  fill(vox, 0u, 384*640);
  vox_scatter_kernel<<<150, 256, 0, stream>>>(mee_vidx, mee_val, vox);
  gemm(vox, PAR(27), PAR(28), nullptr, mee_t, 384, 64, 640, 1, 2); // mee_t = gelu(vox @ c1_w^T + b)

  // ---- instance-norm-1 stats (analytic for mae_r via Gram matrix)
  in1_stats_mee_kernel<<<1, 64, 0, stream>>>(mee_t, mean1, inv1);
  fill(G, 0u, 1024);
  fill(Svec, 0u, 32);
  mae_gram_kernel<<<64, 256, 0, stream>>>(mae, G, Svec);
  in1_stats_mae_kernel<<<1, 128, 0, stream>>>(G, Svec, PAR(29), PAR(30), mean1, inv1);

  // ---- stage2 fused conv1x1 256->128 + stage3 strided conv (WMMA)
  stage2_kernel<<<2304, 256, 0, stream>>>(mae, mee_t, PAR(29), PAR(30), mean1, inv1,
                                          PAR(33), PAR(34), PAR(31), PAR(32), Y);
  in2_stats_kernel<<<128, 256, 0, stream>>>(Y, mean2, inv2);
  stage3_kernel<<<dim3(24,24), 256, 0, stream>>>(Y, mean2, inv2, PAR(35), PAR(36), PAR(37), PAR(38), Xpp);

  // ---- pair2pair: row MHA
  ln_kernel<<<4608, 256, 0, stream>>>(Xpp, PAR(39), PAR(40), Sb, 36864);
  gemm(Sb, PAR(51), PAR(52), nullptr, Qb, 36864, 128, 128, 0, 0);
  gemm(Sb, PAR(53), PAR(54), nullptr, Kb, 36864, 128, 128, 0, 0);
  gemm(Sb, PAR(55), PAR(56), nullptr, Vb, 36864, 128, 128, 0, 0);
  attn_kernel<<<dim3(12,4,192), 32, 0, stream>>>(Qb, Kb, Vb, Ob, 0);
  gemm(Ob, PAR(57), PAR(58), Xpp, Xpp, 36864, 128, 128, 0, 0);
  // ---- col MHA
  ln_kernel<<<4608, 256, 0, stream>>>(Xpp, PAR(41), PAR(42), Sb, 36864);
  gemm(Sb, PAR(59), PAR(60), nullptr, Qb, 36864, 128, 128, 0, 0);
  gemm(Sb, PAR(61), PAR(62), nullptr, Kb, 36864, 128, 128, 0, 0);
  gemm(Sb, PAR(63), PAR(64), nullptr, Vb, 36864, 128, 128, 0, 0);
  attn_kernel<<<dim3(12,4,192), 32, 0, stream>>>(Qb, Kb, Vb, Ob, 1);
  gemm(Ob, PAR(65), PAR(66), Xpp, Xpp, 36864, 128, 128, 0, 0);
  // ---- FF
  ln_kernel<<<4608, 256, 0, stream>>>(Xpp, PAR(43), PAR(44), Sb, 36864);
  gemm(Sb, PAR(45), PAR(46), nullptr, Hb, 36864, 256, 128, 0, 1);
  gemm(Hb, PAR(47), PAR(48), Xpp, Xpp, 36864, 128, 256, 0, 0);

  // ---- final masked mean + sigmoid
  fill(partial, 0u, 1);
  final_reduce_kernel<<<144, 256, 0, stream>>>(Xpp, rep1d, PAR(49), PAR(50), partial);
  final_sigmoid_kernel<<<1, 32, 0, stream>>>(partial, (float*)d_out);
#undef PAR
}
